// EdgeTransformerBlock_65146063946130
// MI455X (gfx1250) — matrix operand order repaired and checked by reading, not compiled, not run on hardware
//
#include <hip/hip_runtime.h>
#include <hip/hip_bf16.h>
#include <math.h>

#define DEV __device__ __forceinline__

typedef __attribute__((ext_vector_type(16))) __bf16 v16bf;
typedef __attribute__((ext_vector_type(8)))  float  v8f;

union FragU { v16bf f; uint4 q[2]; };

constexpr int B  = 8;
constexpr int N  = 384;
constexpr int D  = 512;
constexpr int H  = 8;
constexpr int DE = 32;
constexpr int DF = 2048;
constexpr int DK = D / H;      // 64
constexpr int BN = B * N;      // 3072
constexpr int D2 = 2 * D;      // 1024
constexpr int HDE = H * DE;    // 256

DEV v8f zero8() { v8f z = {0.f,0.f,0.f,0.f,0.f,0.f,0.f,0.f}; return z; }

DEV v8f wmma_bf16(v16bf a, v16bf b, v8f c) {
  // D = A(16x32 bf16) x B(32x16 bf16) + C(f32)
  return __builtin_amdgcn_wmma_f32_16x16x32_bf16(false, a, false, b, (short)0, c, false, false);
}

DEV __bf16 f2bf(float x) {
  union { __hip_bfloat16 s; __bf16 h; } u;
  u.s = __float2bfloat16(x);
  return u.h;
}

// A-fragment (M x K row-major bf16). Lane l: row = row0+(l&15); halves 0..7 hold
// K = k0+kb..k0+kb+7, halves 8..15 hold K = k0+16+kb.. (kb = l<16 ? 0 : 8).
DEV v16bf load_frag_a(const __hip_bfloat16* base, int ld, int row0, int k0) {
  const int lane = threadIdx.x & 31;
  const __hip_bfloat16* p =
      base + (size_t)(row0 + (lane & 15)) * ld + k0 + ((lane >> 4) << 3);
  FragU fu;
  fu.q[0] = *reinterpret_cast<const uint4*>(p);
  fu.q[1] = *reinterpret_cast<const uint4*>(p + 16);
  return fu.f;
}

// B-fragment from Bt stored (Ncols x K) row-major bf16 (i.e. column n of B is
// row n of Bt, contiguous over K). Lane l: col = col0+(l&15); halves 0..7 hold
// K = k0+kb..k0+kb+7, halves 8..15 hold K = k0+16+kb.. (kb = l<16 ? 0 : 8).
DEV v16bf load_frag_b(const __hip_bfloat16* Bt, int ldb, int col0, int k0) {
  const int lane = threadIdx.x & 31;
  const __hip_bfloat16* p =
      Bt + (size_t)(col0 + (lane & 15)) * ldb + k0 + ((lane >> 4) << 3);
  FragU fu;
  fu.q[0] = *reinterpret_cast<const uint4*>(p);
  fu.q[1] = *reinterpret_cast<const uint4*>(p + 16);
  return fu.f;
}

DEV void zero_acc(v8f (&a)[2][2]) {
  a[0][0] = zero8(); a[0][1] = zero8(); a[1][0] = zero8(); a[1][1] = zero8();
}

// 32x32 output tile per wave, register-blocked 2x2 over 16x16 WMMA tiles.
DEV void gemm_tile32(const __hip_bfloat16* A, int lda, int row0,
                     const __hip_bfloat16* Bt, int ldb, int col0,
                     int ksteps, v8f (&acc)[2][2]) {
  for (int ks = 0; ks < ksteps; ++ks) {
    const int k0 = ks * 32;
    v16bf a0 = load_frag_a(A, lda, row0,       k0);
    v16bf a1 = load_frag_a(A, lda, row0 + 16,  k0);
    v16bf b0 = load_frag_b(Bt, ldb, col0,      k0);
    v16bf b1 = load_frag_b(Bt, ldb, col0 + 16, k0);
    acc[0][0] = wmma_bf16(a0, b0, acc[0][0]);
    acc[0][1] = wmma_bf16(a0, b1, acc[0][1]);
    acc[1][0] = wmma_bf16(a1, b0, acc[1][0]);
    acc[1][1] = wmma_bf16(a1, b1, acc[1][1]);
  }
}

// ---------------------------------------------------------------- weights prep
__global__ void k_transpose_bf16(const float* __restrict__ W,
                                 __hip_bfloat16* __restrict__ Wt, int R, int C) {
  int idx = blockIdx.x * blockDim.x + threadIdx.x;
  if (idx >= R * C) return;
  int r = idx / C, c = idx - r * C;
  Wt[(size_t)c * R + r] = __float2bfloat16(W[idx]);
}

// ---------------------------------------------------------------- layernorm
__global__ void k_layernorm_bf16(const float* __restrict__ X,
                                 const float* __restrict__ g,
                                 const float* __restrict__ bb,
                                 __hip_bfloat16* __restrict__ out,
                                 int rows, int cols) {
  int gw = (int)((blockIdx.x * blockDim.x + threadIdx.x) >> 5);
  int lane = threadIdx.x & 31;
  if (gw >= rows) return;
  const float* row = X + (size_t)gw * cols;
  float s = 0.f;
  for (int i = lane; i < cols; i += 32) s += row[i];
#pragma unroll
  for (int o = 16; o > 0; o >>= 1) s += __shfl_xor(s, o, 32);
  float mean = s / cols;
  float v = 0.f;
  for (int i = lane; i < cols; i += 32) { float d = row[i] - mean; v += d * d; }
#pragma unroll
  for (int o = 16; o > 0; o >>= 1) v += __shfl_xor(v, o, 32);
  float inv = rsqrtf(v / cols + 1e-5f);
  __hip_bfloat16* orow = out + (size_t)gw * cols;
  for (int i = lane; i < cols; i += 32)
    orow[i] = __float2bfloat16((row[i] - mean) * inv * g[i] + bb[i]);
}

// ---------------------------------------------------------------- QKV GEMM
__global__ void k_qkv(const __hip_bfloat16* __restrict__ nx,
                      const __hip_bfloat16* __restrict__ WqT,
                      const __hip_bfloat16* __restrict__ WkT,
                      const __hip_bfloat16* __restrict__ WvT,
                      const float* __restrict__ bq, const float* __restrict__ bk,
                      const float* __restrict__ bv,
                      __hip_bfloat16* __restrict__ qh,
                      __hip_bfloat16* __restrict__ kh,
                      __hip_bfloat16* __restrict__ vth) {
  int gw = (int)((blockIdx.x * blockDim.x + threadIdx.x) >> 5);
  const int CT = (3 * D) / 32;                   // 48 col tiles
  int ct = gw % CT, rt = gw / CT;
  int mat = (ct * 32) >> 9;                      // 0=q 1=k 2=v
  int c0 = (ct * 32) & (D - 1);
  const __hip_bfloat16* Bt = (mat == 0) ? WqT : (mat == 1) ? WkT : WvT;
  const float* bias = (mat == 0) ? bq : (mat == 1) ? bk : bv;
  v8f acc[2][2]; zero_acc(acc);
  gemm_tile32(nx, D, rt * 32, Bt, D, c0, D / 32, acc);
  int roff = (threadIdx.x & 16) ? 8 : 0;
  int col  = threadIdx.x & 15;
#pragma unroll
  for (int i = 0; i < 2; ++i)
#pragma unroll
    for (int j = 0; j < 2; ++j)
#pragma unroll
      for (int r = 0; r < 8; ++r) {
        int rg = rt * 32 + i * 16 + roff + r;
        int c  = c0 + j * 16 + col;
        float val = acc[i][j][r] + bias[c];
        int b = rg / N, nn = rg - b * N;
        int h = c >> 6, dk = c & 63;
        __hip_bfloat16 o = __float2bfloat16(val);
        if (mat == 2)      vth[(((size_t)(b * H + h)) * DK + dk) * N + nn] = o;
        else if (mat == 1) kh [(((size_t)(b * H + h)) * N + nn) * DK + dk] = o;
        else               qh [(((size_t)(b * H + h)) * N + nn) * DK + dk] = o;
      }
}

// ---------------- edge LN + Wke projection (stored transposed) + scalar bias
__global__ void k_edge(const float* __restrict__ E,
                       const float* __restrict__ lg, const float* __restrict__ lb,
                       const __hip_bfloat16* __restrict__ WkeT,
                       const float* __restrict__ bke,
                       const float* __restrict__ Web, const float* __restrict__ beb,
                       __hip_bfloat16* __restrict__ kedgeT,
                       float* __restrict__ biasO) {
  int gw = (int)((blockIdx.x * blockDim.x + threadIdx.x) >> 5);
  int lane = threadIdx.x & 31;
  const int MT = N / 16;
  int mt = gw % MT, t = gw / MT;
  int n = t % N, b = t / N;
  int m  = mt * 16 + (lane & 15);
  int kb = (lane & 16) ? 8 : 0;
  const float* rowp = E + (((size_t)b * N + n) * N + m) * DE;
  float vals[16];
#pragma unroll
  for (int j = 0; j < 8; ++j) { vals[j] = rowp[kb + j]; vals[8 + j] = rowp[kb + 16 + j]; }
  float s = 0.f;
#pragma unroll
  for (int j = 0; j < 16; ++j) s += vals[j];
  s += __shfl_xor(s, 16, 32);
  float mean = s * (1.f / 32.f);
  float var = 0.f;
#pragma unroll
  for (int j = 0; j < 16; ++j) { float d = vals[j] - mean; var += d * d; }
  var += __shfl_xor(var, 16, 32);
  float inv = rsqrtf(var * (1.f / 32.f) + 1e-5f);
  FragU fa;
  float bp = 0.f;
#pragma unroll
  for (int j = 0; j < 8; ++j) {
    int cA = kb + j, cB = kb + 16 + j;
    float x0 = (vals[j]     - mean) * inv * lg[cA] + lb[cA];
    float x1 = (vals[8 + j] - mean) * inv * lg[cB] + lb[cB];
    fa.f[j]     = f2bf(x0);
    fa.f[8 + j] = f2bf(x1);
    bp += x0 * Web[cA] + x1 * Web[cB];
  }
  bp += __shfl_xor(bp, 16, 32);
  v16bf b0 = load_frag_b(WkeT, DE, 0,  0);
  v16bf b1 = load_frag_b(WkeT, DE, 16, 0);
  v8f c0 = zero8(), c1 = zero8();
  c0 = wmma_bf16(fa.f, b0, c0);
  c1 = wmma_bf16(fa.f, b1, c1);
  int roff = (lane & 16) ? 8 : 0;
  int col  = lane & 15;
  __hip_bfloat16* kbase = kedgeT + ((size_t)b * N + n) * DE * N;
#pragma unroll
  for (int r = 0; r < 8; ++r) {
    int mm = mt * 16 + roff + r;
    kbase[(size_t)col * N + mm]        = __float2bfloat16(c0[r] + bke[col]);
    kbase[(size_t)(col + 16) * N + mm] = __float2bfloat16(c1[r] + bke[col + 16]);
  }
  if (lane < 16)
    biasO[((size_t)b * N + n) * N + m] = (bp + beb[0]) * 0.70710678118f;
}

// -------------------- scores + two-pass online softmax -> attn (bf16)
__global__ void k_scores(const __hip_bfloat16* __restrict__ qh,
                         const __hip_bfloat16* __restrict__ kh,
                         const float* __restrict__ biasO,
                         const unsigned char* __restrict__ mask,
                         __hip_bfloat16* __restrict__ attn) {
  int gw = (int)((blockIdx.x * blockDim.x + threadIdx.x) >> 5);
  const int NT = N / 16;
  int nt = gw % NT, bh = gw / NT;
  int b = bh >> 3;
  const __hip_bfloat16* qb = qh + (size_t)bh * N * DK;
  const __hip_bfloat16* kb = kh + (size_t)bh * N * DK;
  int n0 = nt * 16;
  int lane = threadIdx.x & 31;
  int roff = (lane & 16) ? 8 : 0;
  int col  = lane & 15;
  const float scale = 0.08838834764f;  // (2*DK)^-0.5
  v16bf a0 = load_frag_a(qb, DK, n0, 0);
  v16bf a1 = load_frag_a(qb, DK, n0, 32);
  float rmax[8], rsum[8];
#pragma unroll
  for (int r = 0; r < 8; ++r) { rmax[r] = -3.0e38f; rsum[r] = 0.f; }
  for (int mt = 0; mt < NT; ++mt) {
    v16bf b0 = load_frag_b(kb, DK, mt * 16, 0);
    v16bf b1 = load_frag_b(kb, DK, mt * 16, 32);
    v8f acc = zero8();
    acc = wmma_bf16(a0, b0, acc);
    acc = wmma_bf16(a1, b1, acc);
    int mcol = mt * 16 + col;
#pragma unroll
    for (int r = 0; r < 8; ++r) {
      int nrow = n0 + roff + r;
      size_t bi = ((size_t)b * N + nrow) * N + mcol;
      float sv = acc[r] * scale + biasO[bi];
      if (mask[bi]) sv = -3.0e38f;
      float mo = rmax[r];
      float mn = fmaxf(mo, sv);
      rsum[r] = rsum[r] * __expf(mo - mn) + __expf(sv - mn);
      rmax[r] = mn;
    }
  }
#pragma unroll
  for (int o = 1; o < 16; o <<= 1) {
#pragma unroll
    for (int r = 0; r < 8; ++r) {
      float mo = __shfl_xor(rmax[r], o, 32);
      float so = __shfl_xor(rsum[r], o, 32);
      float mn = fmaxf(rmax[r], mo);
      rsum[r] = rsum[r] * __expf(rmax[r] - mn) + so * __expf(mo - mn);
      rmax[r] = mn;
    }
  }
  float rinv[8];
#pragma unroll
  for (int r = 0; r < 8; ++r) rinv[r] = 1.f / rsum[r];
  for (int mt = 0; mt < NT; ++mt) {
    v16bf b0 = load_frag_b(kb, DK, mt * 16, 0);
    v16bf b1 = load_frag_b(kb, DK, mt * 16, 32);
    v8f acc = zero8();
    acc = wmma_bf16(a0, b0, acc);
    acc = wmma_bf16(a1, b1, acc);
    int mcol = mt * 16 + col;
#pragma unroll
    for (int r = 0; r < 8; ++r) {
      int nrow = n0 + roff + r;
      size_t bi = ((size_t)b * N + nrow) * N + mcol;
      float sv = acc[r] * scale + biasO[bi];
      if (mask[bi]) sv = -3.0e38f;
      float p = __expf(sv - rmax[r]) * rinv[r];
      attn[((size_t)bh * N + nrow) * N + mcol] = __float2bfloat16(p);
    }
  }
}

// -------------------- ctx = attn @ v  -> cat[:, 0:512]
__global__ void k_ctx(const __hip_bfloat16* __restrict__ attn,
                      const __hip_bfloat16* __restrict__ vth,
                      __hip_bfloat16* __restrict__ cat) {
  int gw = (int)((blockIdx.x * blockDim.x + threadIdx.x) >> 5);
  int dt = gw % 2;
  int nt = (gw / 2) % (N / 32);
  int bh = gw / (2 * (N / 32));
  int b = bh >> 3, h = bh & 7;
  const __hip_bfloat16* A  = attn + (size_t)bh * N * N;
  const __hip_bfloat16* Bt = vth + (size_t)bh * DK * N;
  v8f acc[2][2]; zero_acc(acc);
  gemm_tile32(A, N, nt * 32, Bt, N, dt * 32, N / 32, acc);
  int roff = (threadIdx.x & 16) ? 8 : 0;
  int col  = threadIdx.x & 15;
#pragma unroll
  for (int i = 0; i < 2; ++i)
#pragma unroll
    for (int j = 0; j < 2; ++j)
#pragma unroll
      for (int r = 0; r < 8; ++r) {
        int n  = nt * 32 + i * 16 + roff + r;
        int dk = dt * 32 + j * 16 + col;
        cat[((size_t)(b * N + n)) * D2 + h * DK + dk] =
            __float2bfloat16(acc[i][j][r]);
      }
}

// -------------------- ectx[h,e] = sum_m attn[b,h,n,m] * k_edge[b,n,m,e]
__global__ void k_ectx(const __hip_bfloat16* __restrict__ attn,
                       const __hip_bfloat16* __restrict__ kedgeT,
                       __hip_bfloat16* __restrict__ ectx) {
  int gw = (int)((blockIdx.x * blockDim.x + threadIdx.x) >> 5);
  int lane = threadIdx.x & 31;
  int n = gw % N, b = gw / N;
  int hrow = lane & 15;
  int kb = (lane & 16) ? 8 : 0;
  bool valid = hrow < H;
  const __hip_bfloat16* ap =
      attn + (((size_t)(b * H + (hrow & 7)) * N + n) * N);
  const __hip_bfloat16* bp = kedgeT + ((size_t)b * N + n) * DE * N;
  uint4 z; z.x = z.y = z.z = z.w = 0u;
  v8f c0 = zero8(), c1 = zero8();
  for (int ks = 0; ks < N / 32; ++ks) {
    int k0 = ks * 32 + kb;
    FragU fa;
    fa.q[0] = valid ? *reinterpret_cast<const uint4*>(ap + k0)      : z;
    fa.q[1] = valid ? *reinterpret_cast<const uint4*>(ap + k0 + 16) : z;
    v16bf b0 = load_frag_b(bp, N, 0,  ks * 32);
    v16bf b1 = load_frag_b(bp, N, 16, ks * 32);
    c0 = wmma_bf16(fa.f, b0, c0);
    c1 = wmma_bf16(fa.f, b1, c1);
  }
  int roff = (lane & 16) ? 8 : 0;
  int col  = lane & 15;
  __hip_bfloat16* ob = ectx + ((size_t)(b * N + n)) * HDE;
#pragma unroll
  for (int r = 0; r < 8; ++r) {
    int hh = roff + r;
    if (hh < H) {
      ob[hh * DE + col]      = __float2bfloat16(c0[r]);
      ob[hh * DE + 16 + col] = __float2bfloat16(c1[r]);
    }
  }
}

// -------------------- ectx @ Weo + beo -> cat[:, 512:1024]
__global__ void k_ectx_proj(const __hip_bfloat16* __restrict__ ectx,
                            const __hip_bfloat16* __restrict__ WeoT,
                            const float* __restrict__ beo,
                            __hip_bfloat16* __restrict__ cat) {
  int gw = (int)((blockIdx.x * blockDim.x + threadIdx.x) >> 5);
  const int CT = D / 32;
  int ct = gw % CT, rt = gw / CT;
  v8f acc[2][2]; zero_acc(acc);
  gemm_tile32(ectx, HDE, rt * 32, WeoT, HDE, ct * 32, HDE / 32, acc);
  int roff = (threadIdx.x & 16) ? 8 : 0;
  int col  = threadIdx.x & 15;
#pragma unroll
  for (int i = 0; i < 2; ++i)
#pragma unroll
    for (int j = 0; j < 2; ++j)
#pragma unroll
      for (int r = 0; r < 8; ++r) {
        int rr = rt * 32 + i * 16 + roff + r;
        int c  = ct * 32 + j * 16 + col;
        cat[(size_t)rr * D2 + D + c] = __float2bfloat16(acc[i][j][r] + beo[c]);
      }
}

// -------------------- cat @ Wo + bo + x -> x1 (fp32 residual)
__global__ void k_attn_out(const __hip_bfloat16* __restrict__ cat,
                           const __hip_bfloat16* __restrict__ WoT,
                           const float* __restrict__ bo,
                           const float* __restrict__ x,
                           float* __restrict__ x1) {
  int gw = (int)((blockIdx.x * blockDim.x + threadIdx.x) >> 5);
  const int CT = D / 32;
  int ct = gw % CT, rt = gw / CT;
  v8f acc[2][2]; zero_acc(acc);
  gemm_tile32(cat, D2, rt * 32, WoT, D2, ct * 32, D2 / 32, acc);
  int roff = (threadIdx.x & 16) ? 8 : 0;
  int col  = threadIdx.x & 15;
#pragma unroll
  for (int i = 0; i < 2; ++i)
#pragma unroll
    for (int j = 0; j < 2; ++j)
#pragma unroll
      for (int r = 0; r < 8; ++r) {
        int rr = rt * 32 + i * 16 + roff + r;
        int c  = ct * 32 + j * 16 + col;
        size_t o = (size_t)rr * D + c;
        x1[o] = acc[i][j][r] + bo[c] + x[o];
      }
}

// -------------------- FFN1 with gelu -> bf16
__global__ void k_ffn1(const __hip_bfloat16* __restrict__ h2,
                       const __hip_bfloat16* __restrict__ W1T,
                       const float* __restrict__ b1,
                       __hip_bfloat16* __restrict__ f1) {
  int gw = (int)((blockIdx.x * blockDim.x + threadIdx.x) >> 5);
  const int CT = DF / 32;
  int ct = gw % CT, rt = gw / CT;
  v8f acc[2][2]; zero_acc(acc);
  gemm_tile32(h2, D, rt * 32, W1T, D, ct * 32, D / 32, acc);
  int roff = (threadIdx.x & 16) ? 8 : 0;
  int col  = threadIdx.x & 15;
#pragma unroll
  for (int i = 0; i < 2; ++i)
#pragma unroll
    for (int j = 0; j < 2; ++j)
#pragma unroll
      for (int r = 0; r < 8; ++r) {
        int rr = rt * 32 + i * 16 + roff + r;
        int c  = ct * 32 + j * 16 + col;
        float t = acc[i][j][r] + b1[c];
        float g = 0.5f * t * (1.f + tanhf(0.7978845608f * (t + 0.044715f * t * t * t)));
        f1[(size_t)rr * DF + c] = __float2bfloat16(g);
      }
}

// -------------------- FFN2 + residual -> out (fp32)
__global__ void k_ffn2(const __hip_bfloat16* __restrict__ f1,
                       const __hip_bfloat16* __restrict__ W2T,
                       const float* __restrict__ b2,
                       const float* __restrict__ x1,
                       float* __restrict__ out) {
  int gw = (int)((blockIdx.x * blockDim.x + threadIdx.x) >> 5);
  const int CT = D / 32;
  int ct = gw % CT, rt = gw / CT;
  v8f acc[2][2]; zero_acc(acc);
  gemm_tile32(f1, DF, rt * 32, W2T, DF, ct * 32, DF / 32, acc);
  int roff = (threadIdx.x & 16) ? 8 : 0;
  int col  = threadIdx.x & 15;
#pragma unroll
  for (int i = 0; i < 2; ++i)
#pragma unroll
    for (int j = 0; j < 2; ++j)
#pragma unroll
      for (int r = 0; r < 8; ++r) {
        int rr = rt * 32 + i * 16 + roff + r;
        int c  = ct * 32 + j * 16 + col;
        size_t o = (size_t)rr * D + c;
        out[o] = acc[i][j][r] + b2[c] + x1[o];
      }
}

// ================================================================ host launch
extern "C" void kernel_launch(void* const* d_in, const int* in_sizes, int n_in,
                              void* d_out, int out_size, void* d_ws, size_t ws_size,
                              hipStream_t stream) {
  const float* x    = (const float*)d_in[0];
  const float* E    = (const float*)d_in[1];
  const unsigned char* mask = (const unsigned char*)d_in[2];
  const float* ln_a_g = (const float*)d_in[3];
  const float* ln_a_b = (const float*)d_in[4];
  const float* ln_e_g = (const float*)d_in[5];
  const float* ln_e_b = (const float*)d_in[6];
  const float* Wq = (const float*)d_in[7];   const float* bq  = (const float*)d_in[8];
  const float* Wk = (const float*)d_in[9];   const float* bk  = (const float*)d_in[10];
  const float* Wv = (const float*)d_in[11];  const float* bv  = (const float*)d_in[12];
  const float* Wke = (const float*)d_in[13]; const float* bke = (const float*)d_in[14];
  const float* Web = (const float*)d_in[15]; const float* beb = (const float*)d_in[16];
  const float* Weo = (const float*)d_in[17]; const float* beo = (const float*)d_in[18];
  const float* Wo  = (const float*)d_in[19]; const float* bo  = (const float*)d_in[20];
  const float* ln_f_g = (const float*)d_in[21];
  const float* ln_f_b = (const float*)d_in[22];
  const float* W1 = (const float*)d_in[23];  const float* b1f = (const float*)d_in[24];
  const float* W2 = (const float*)d_in[25];  const float* b2f = (const float*)d_in[26];
  float* out = (float*)d_out;

  char* w = (char*)d_ws;
  size_t off = 0;
  auto alloc = [&](size_t bytes) -> void* {
    void* p = w + off;
    off = (off + bytes + 255) & ~(size_t)255;
    return p;
  };

  __hip_bfloat16* nx    = (__hip_bfloat16*)alloc((size_t)BN * D * 2);
  __hip_bfloat16* WqT   = (__hip_bfloat16*)alloc((size_t)D * D * 2);
  __hip_bfloat16* WkT   = (__hip_bfloat16*)alloc((size_t)D * D * 2);
  __hip_bfloat16* WvT   = (__hip_bfloat16*)alloc((size_t)D * D * 2);
  __hip_bfloat16* WkeT  = (__hip_bfloat16*)alloc((size_t)DE * DE * 2);
  __hip_bfloat16* WeoT  = (__hip_bfloat16*)alloc((size_t)D * HDE * 2);
  __hip_bfloat16* WoT   = (__hip_bfloat16*)alloc((size_t)D * D2 * 2);
  __hip_bfloat16* W1T   = (__hip_bfloat16*)alloc((size_t)DF * D * 2);
  __hip_bfloat16* W2T   = (__hip_bfloat16*)alloc((size_t)D * DF * 2);
  __hip_bfloat16* qh    = (__hip_bfloat16*)alloc((size_t)B * H * N * DK * 2);
  __hip_bfloat16* kh    = (__hip_bfloat16*)alloc((size_t)B * H * N * DK * 2);
  __hip_bfloat16* vth   = (__hip_bfloat16*)alloc((size_t)B * H * DK * N * 2);
  __hip_bfloat16* kedgeT= (__hip_bfloat16*)alloc((size_t)B * N * DE * N * 2);
  float*          biasO = (float*)alloc((size_t)B * N * N * 4);
  __hip_bfloat16* attn  = (__hip_bfloat16*)alloc((size_t)B * H * N * N * 2);
  __hip_bfloat16* ectx  = (__hip_bfloat16*)alloc((size_t)BN * HDE * 2);
  __hip_bfloat16* cat   = (__hip_bfloat16*)alloc((size_t)BN * D2 * 2);
  float*          x1    = (float*)alloc((size_t)BN * D * 4);
  __hip_bfloat16* h2    = (__hip_bfloat16*)alloc((size_t)BN * D * 2);
  __hip_bfloat16* f1    = (__hip_bfloat16*)alloc((size_t)BN * DF * 2);
  (void)ws_size; (void)in_sizes; (void)n_in; (void)out_size;

  auto T = [&](const float* Wp, __hip_bfloat16* Wt, int R, int C) {
    int n = R * C;
    k_transpose_bf16<<<(n + 255) / 256, 256, 0, stream>>>(Wp, Wt, R, C);
  };
  T(Wq, WqT, D, D);     T(Wk, WkT, D, D);     T(Wv, WvT, D, D);
  T(Wke, WkeT, DE, DE); T(Weo, WeoT, HDE, D); T(Wo, WoT, D2, D);
  T(W1, W1T, D, DF);    T(W2, W2T, DF, D);

  k_layernorm_bf16<<<BN / 8, 256, 0, stream>>>(x, ln_a_g, ln_a_b, nx, BN, D);

  k_qkv<<<((BN / 32) * ((3 * D) / 32)) / 8, 256, 0, stream>>>(
      nx, WqT, WkT, WvT, bq, bk, bv, qh, kh, vth);

  k_edge<<<(B * N * (N / 16)) / 8, 256, 0, stream>>>(
      E, ln_e_g, ln_e_b, WkeT, bke, Web, beb, kedgeT, biasO);

  k_scores<<<(B * H * (N / 16)) / 8, 256, 0, stream>>>(
      qh, kh, biasO, mask, attn);

  k_ctx<<<(B * H * (N / 32) * 2) / 8, 256, 0, stream>>>(attn, vth, cat);

  k_ectx<<<(B * N) / 8, 256, 0, stream>>>(attn, kedgeT, ectx);

  k_ectx_proj<<<((BN / 32) * (D / 32)) / 8, 256, 0, stream>>>(
      ectx, WeoT, beo, cat);

  k_attn_out<<<((BN / 32) * (D / 32)) / 8, 256, 0, stream>>>(
      cat, WoT, bo, x, x1);

  k_layernorm_bf16<<<BN / 8, 256, 0, stream>>>(x1, ln_f_g, ln_f_b, h2, BN, D);

  k_ffn1<<<((BN / 32) * (DF / 32)) / 8, 256, 0, stream>>>(h2, W1T, b1f, f1);

  k_ffn2<<<((BN / 32) * (D / 32)) / 8, 256, 0, stream>>>(f1, W2T, b2f, x1, out);
}
